// TtCausalSelfAttention_41455024341160
// MI455X (gfx1250) — matrix operand order repaired and checked by reading, not compiled, not run on hardware
//
#include <hip/hip_runtime.h>

// GPT-2 small attention: B=8, T=1024, C=768, H=12, D=64
#define BATCH   8
#define SEQ_T   1024
#define CDIM    768
#define H_HEADS 12
#define DHEAD   64
#define QKV_LD  2304   // 3*C

typedef __attribute__((ext_vector_type(16))) _Float16     v16h;
typedef __attribute__((ext_vector_type(8)))  _Float16     v8h;
typedef __attribute__((ext_vector_type(8)))  float        v8f;
typedef __attribute__((ext_vector_type(4)))  unsigned int v4ui;
typedef __attribute__((ext_vector_type(8)))  int          v8si;
typedef __attribute__((ext_vector_type(4)))  int          v4si;

__device__ __forceinline__ v8f wmma_f16f32(v16h a, v16h b, v8f c) {
  // D = A(16x32 f16) * B(32x16 f16) + C(16x16 f32)
  return __builtin_amdgcn_wmma_f32_16x16x32_f16(false, a, false, b, (short)0, c,
                                                false, false);
}

__device__ __forceinline__ v8f vzero8() {
  v8f z = {0.f, 0.f, 0.f, 0.f, 0.f, 0.f, 0.f, 0.f};
  return z;
}

// A fragment 16x32 (MxK) from row-major f16 with leading dim lda.
// ISA layout: lanes 0-15 row M=lane, K 0-7 then 16-23; lanes 16-31 same rows,
// K 8-15 then 24-31. Two 16B loads per lane.
__device__ __forceinline__ v16h load_frag_a(const _Float16* a, int lda) {
  const int lane = threadIdx.x & 31;
  const int row  = lane & 15;
  const int k0   = (lane >> 4) * 8;
  const _Float16* p = a + row * lda + k0;
  v16h r;
  *(v8h*)&r         = *(const v8h*)(p);
  *(((v8h*)&r) + 1) = *(const v8h*)(p + 16);
  return r;
}

// B fragment 32x16 (KxN) where storage is K-contiguous per column:
// bt[n*ld + k].  Lanes 0-15: col N=lane, K 0-7 then 16-23; lanes 16-31:
// col N=lane-16, K 8-15 then 24-31.  Two 16B loads per lane.
__device__ __forceinline__ v16h load_frag_b_kcont(const _Float16* bt, int ld) {
  const int lane = threadIdx.x & 31;
  const int col  = lane & 15;
  const int kb   = (lane >> 4) * 8;
  const _Float16* p = bt + col * ld + kb;
  v16h r;
  *(v8h*)&r         = *(const v8h*)(p);
  *(((v8h*)&r) + 1) = *(const v8h*)(p + 16);
  return r;
}

// ---------------------------------------------------------------------------
// Tensor Data Mover: 2D tile load Global -> LDS (CDNA5 ISA
// sections 8.3/8.4).  D# group0: count=1, lds_addr, global_addr, type=2.
// D# group1: data_size=1 (2B), tensor_dim0/1, tile_dim0/1, dim0 stride.
// ---------------------------------------------------------------------------
#if __has_builtin(__builtin_amdgcn_tensor_load_to_lds)
#define HAVE_TDM 1
__device__ __forceinline__ void tdm_load_2d(const _Float16* gsrc,
                                            unsigned lds_off,
                                            unsigned tile_d0,   // elems/row
                                            unsigned tile_d1,   // rows
                                            unsigned stride_el) // row stride
{
  const unsigned long long ga = (unsigned long long)(size_t)gsrc;
  v4ui g0;
  g0[0] = 1u;                                    // count=1 (valid user D#)
  g0[1] = lds_off;                               // lds_addr (bytes)
  g0[2] = (unsigned)ga;                          // global_addr[31:0]
  g0[3] = (unsigned)((ga >> 32) & 0x01FFFFFFu)   // global_addr[56:32]
          | 0x80000000u;                         // type = 2 ("image")
  const unsigned td0 = stride_el;                // tensor_dim0 (elems)
  const unsigned td1 = 1u << 20;                 // tensor_dim1 (ample rows)
  v8si g1;
  g1[0] = (int)(1u << 16);                                   // data_size=1 (2B)
  g1[1] = (int)((td0 & 0xFFFFu) << 16);                      // dim0[15:0]
  g1[2] = (int)(((td0 >> 16) & 0xFFFFu) | ((td1 & 0xFFFFu) << 16));
  g1[3] = (int)(((td1 >> 16) & 0xFFFFu) | ((tile_d0 & 0xFFFFu) << 16));
  g1[4] = (int)(tile_d1 & 0xFFFFu);                          // tile_dim1, dim2=0
  g1[5] = (int)stride_el;                                    // dim0_stride lo
  g1[6] = 0;
  g1[7] = 0;
  v4si z4 = {0, 0, 0, 0};
#if __has_include(<hip/amd_detail/amd_gfx1250_TDM.h>)
  v8si z8 = {0, 0, 0, 0, 0, 0, 0, 0};
  __builtin_amdgcn_tensor_load_to_lds(g0, g1, z4, z4, z8, 0);
#else
  __builtin_amdgcn_tensor_load_to_lds(g0, g1, z4, z4, 0);
#endif
}
#else
#define HAVE_TDM 0
#endif

// ---------------------------------------------------------------------------
// f32 -> f16 conversion / transpose kernels (memory-bound, tiny vs GEMM work)
// ---------------------------------------------------------------------------
__global__ __launch_bounds__(256) void cvt_f32_f16(const float* __restrict__ s,
                                                   _Float16* __restrict__ d,
                                                   int n) {
  int i = blockIdx.x * blockDim.x + threadIdx.x;
  const int stride = gridDim.x * blockDim.x;
  for (; i < n; i += stride) d[i] = (_Float16)s[i];
}

// wt[n*K + k] = (f16) w[k*N + n]   (K-contiguous weight for B fragments)
__global__ __launch_bounds__(256) void transpose_cvt_f32_f16(
    const float* __restrict__ w, _Float16* __restrict__ wt, int K, int N) {
  int i = blockIdx.x * blockDim.x + threadIdx.x;
  const int total = K * N;
  const int stride = gridDim.x * blockDim.x;
  for (; i < total; i += stride) {
    const int n = i / K;
    const int k = i - n * K;
    wt[i] = (_Float16)w[(size_t)k * N + n];
  }
}

// vt[((b*H + h)*D + d)*T + t] = qkv[(b*T + t)*3C + 2C + h*D + d]
// Makes PV B-fragments key-contiguous (two 16B loads/lane instead of 16 b16s).
__global__ __launch_bounds__(256) void transpose_v(
    const _Float16* __restrict__ qkv, _Float16* __restrict__ vt) {
  int i = blockIdx.x * blockDim.x + threadIdx.x;
  const int total = BATCH * CDIM * SEQ_T;
  const int stride = gridDim.x * blockDim.x;
  for (; i < total; i += stride) {
    const int t  = i & (SEQ_T - 1);
    const int d  = (i >> 10) & (DHEAD - 1);
    const int bh = i >> 16;
    const int b  = bh / H_HEADS;
    const int h  = bh - b * H_HEADS;
    vt[i] = qkv[((size_t)(b * SEQ_T + t)) * QKV_LD + 2 * CDIM + h * DHEAD + d];
  }
}

// ---------------------------------------------------------------------------
// GEMM: C[M,N] = A[M,K] * B[K,N] + bias, A f16 row-major, B given as Bt[N,K].
// One wave computes a 16x64 tile.  All 4 waves in a block share the same
// 16-row A tile (n64 % 4 == 0), so the A k-slice is staged in LDS once per
// block via the Tensor Data Mover and read back as ds_load fragments.
// ---------------------------------------------------------------------------
template <bool OUT_F32>
__global__ __launch_bounds__(128) void gemm16x64_wmma(
    const _Float16* __restrict__ A, const _Float16* __restrict__ Bt,
    const float* __restrict__ bias, _Float16* __restrict__ Ch,
    float* __restrict__ Cf, int N, int K, int n64) {
  __shared__ __align__(16) _Float16 atile[16 * 32];

  const int w   = threadIdx.x >> 5;
  const int wid = blockIdx.x * 4 + w;
  const int mt = wid / n64;
  const int nt = wid - mt * n64;
  const int m0 = mt * 16;
  const int n0 = nt * 64;

  v8f acc0 = vzero8(), acc1 = vzero8(), acc2 = vzero8(), acc3 = vzero8();

  const _Float16* Ap  = A + (size_t)m0 * K;
  const _Float16* Bp0 = Bt + (size_t)(n0 + 0)  * K;
  const _Float16* Bp1 = Bt + (size_t)(n0 + 16) * K;
  const _Float16* Bp2 = Bt + (size_t)(n0 + 32) * K;
  const _Float16* Bp3 = Bt + (size_t)(n0 + 48) * K;

#if HAVE_TDM
  const unsigned lds_off = (unsigned)(size_t)(void*)atile;
#endif

  for (int k = 0; k < K; k += 32) {
#if HAVE_TDM
    __syncthreads();                       // previous-iteration reads done
    if (w == 0) {                          // wave-uniform; TDM ignores EXEC
      tdm_load_2d(Ap + k, lds_off, /*tile_d0=*/32, /*tile_d1=*/16,
                  /*stride=*/(unsigned)K);
      __builtin_amdgcn_s_wait_tensorcnt(0);
    }
    __syncthreads();                       // A tile visible to all waves
    const v16h a = load_frag_a(atile, 32); // ds_load_b128 x2 per lane
#else
    const v16h a = load_frag_a(Ap + k, K);
#endif
    if (k + 32 < K) {
      __builtin_prefetch(Bp0 + k + 32, 0, 1);
      __builtin_prefetch(Bp2 + k + 32, 0, 1);
    }
    acc0 = wmma_f16f32(a, load_frag_b_kcont(Bp0 + k, K), acc0);
    acc1 = wmma_f16f32(a, load_frag_b_kcont(Bp1 + k, K), acc1);
    acc2 = wmma_f16f32(a, load_frag_b_kcont(Bp2 + k, K), acc2);
    acc3 = wmma_f16f32(a, load_frag_b_kcont(Bp3 + k, K), acc3);
  }

  const int lane = threadIdx.x & 31;
  const int cn = lane & 15;
  const int hi = lane >> 4;
  v8f accs[4] = {acc0, acc1, acc2, acc3};
#pragma unroll
  for (int t = 0; t < 4; ++t) {
    const int col = n0 + t * 16 + cn;
    const float bv = bias[col];
#pragma unroll
    for (int r = 0; r < 8; ++r) {
      const int row = m0 + hi * 8 + r;
      const float v = accs[t][r] + bv;
      if (OUT_F32) Cf[(size_t)row * N + col] = v;
      else         Ch[(size_t)row * N + col] = (_Float16)v;
    }
  }
}

// ---------------------------------------------------------------------------
// Causal flash attention.  One wave owns a 16-row query tile of one (b,h).
// Streams 32-key blocks: S = Q*K^T via WMMA, online softmax (shfl width-16
// row reductions), P transposed C-layout -> A-layout through per-wave LDS,
// O += P*V via WMMA with V pre-transposed to [B,H,D,T] (key-contiguous).
// No divergent branches around WMMA (EXEC all-1s).
// ---------------------------------------------------------------------------
__global__ __launch_bounds__(128) void flash_attn_wmma(
    const _Float16* __restrict__ qkv,  // [B*T, 3C] f16
    const _Float16* __restrict__ vt,   // [B,H,D,T] f16
    _Float16* __restrict__ y) {        // [B*T, C]  f16
  __shared__ __align__(16) _Float16 plds[4][16 * 32];

  const int w    = threadIdx.x >> 5;
  const int lane = threadIdx.x & 31;
  const int cn   = lane & 15;
  const int hi   = lane >> 4;

  const int wid   = blockIdx.x * 4 + w;
  const int bh    = wid >> 6;          // 64 query tiles per (b,h)
  const int qt    = wid & 63;
  const int b     = bh / H_HEADS;
  const int h     = bh - b * H_HEADS;
  const int qbase = qt * 16;

  const size_t rowbase = (size_t)b * SEQ_T;
  const _Float16* Qb = qkv + (rowbase + qbase) * QKV_LD + h * DHEAD;
  const _Float16* Kb = qkv + rowbase * QKV_LD + CDIM + h * DHEAD;
  const _Float16* Vt = vt + (size_t)(b * H_HEADS + h) * DHEAD * SEQ_T;

  const v16h aq0 = load_frag_a(Qb, QKV_LD);        // Q k=0..31
  const v16h aq1 = load_frag_a(Qb + 32, QKV_LD);   // Q k=32..63

  float m[8], l[8];
  v8f o0 = vzero8(), o1 = vzero8(), o2 = vzero8(), o3 = vzero8();
#pragma unroll
  for (int r = 0; r < 8; ++r) { m[r] = -1e30f; l[r] = 0.f; }

  const int nkb = (qt >> 1) + 1;  // 32-key blocks covering keys <= qbase+15
  for (int kb = 0; kb < nkb; ++kb) {
    const int n0 = kb * 32;

    // S tiles (16 queries x 16 keys each); K row-major is already the
    // K-contiguous B^T layout for Q*K^T.
    v8f s0 = vzero8(), s1 = vzero8();
    {
      const _Float16* K0 = Kb + (size_t)n0 * QKV_LD;
      const _Float16* K1 = Kb + (size_t)(n0 + 16) * QKV_LD;
      s0 = wmma_f16f32(aq0, load_frag_b_kcont(K0,      QKV_LD), s0);
      s0 = wmma_f16f32(aq1, load_frag_b_kcont(K0 + 32, QKV_LD), s0);
      s1 = wmma_f16f32(aq0, load_frag_b_kcont(K1,      QKV_LD), s1);
      s1 = wmma_f16f32(aq1, load_frag_b_kcont(K1 + 32, QKV_LD), s1);
    }

    const float scale = 0.125f;  // 1/sqrt(64)
#pragma unroll
    for (int r = 0; r < 8; ++r) {
      const int qrow = qbase + hi * 8 + r;
      const int key0 = n0 + cn;
      const int key1 = n0 + 16 + cn;
      float v0 = s0[r] * scale; if (key0 > qrow) v0 = -1e30f;  // causal mask
      float v1 = s1[r] * scale; if (key1 > qrow) v1 = -1e30f;

      float rm = fmaxf(v0, v1);
#pragma unroll
      for (int off = 1; off < 16; off <<= 1)
        rm = fmaxf(rm, __shfl_xor(rm, off, 16));
      const float mn = fmaxf(m[r], rm);
      const float p0 = __expf(v0 - mn);
      const float p1 = __expf(v1 - mn);
      float rs = p0 + p1;
#pragma unroll
      for (int off = 1; off < 16; off <<= 1)
        rs += __shfl_xor(rs, off, 16);
      const float alpha = __expf(m[r] - mn);
      l[r] = l[r] * alpha + rs;
      m[r] = mn;
      o0[r] *= alpha; o1[r] *= alpha; o2[r] *= alpha; o3[r] *= alpha;

      // P (C layout) -> LDS 16x32 tile for A-layout reload
      plds[w][(hi * 8 + r) * 32 + cn]      = (_Float16)p0;
      plds[w][(hi * 8 + r) * 32 + 16 + cn] = (_Float16)p1;
    }

    asm volatile("s_wait_dscnt 0" ::: "memory");
    const v16h ap = load_frag_a(&plds[w][0], 32);  // P as 16x32 A fragment
    asm volatile("" ::: "memory");

    // O += P * V, V^T is key-contiguous: two 16B loads per lane per tile.
    o0 = wmma_f16f32(ap, load_frag_b_kcont(Vt + (size_t)0  * SEQ_T + n0, SEQ_T), o0);
    o1 = wmma_f16f32(ap, load_frag_b_kcont(Vt + (size_t)16 * SEQ_T + n0, SEQ_T), o1);
    o2 = wmma_f16f32(ap, load_frag_b_kcont(Vt + (size_t)32 * SEQ_T + n0, SEQ_T), o2);
    o3 = wmma_f16f32(ap, load_frag_b_kcont(Vt + (size_t)48 * SEQ_T + n0, SEQ_T), o3);
  }

  v8f os[4] = {o0, o1, o2, o3};
#pragma unroll
  for (int t = 0; t < 4; ++t) {
#pragma unroll
    for (int r = 0; r < 8; ++r) {
      const size_t row = rowbase + qbase + hi * 8 + r;
      const int col = h * DHEAD + t * 16 + cn;
      y[row * CDIM + col] = (_Float16)(os[t][r] / l[r]);
    }
  }
}

// ---------------------------------------------------------------------------
// Host-side orchestration (stream-ordered, graph-capture safe)
// ---------------------------------------------------------------------------
extern "C" void kernel_launch(void* const* d_in, const int* in_sizes, int n_in,
                              void* d_out, int out_size, void* d_ws,
                              size_t ws_size, hipStream_t stream) {
  (void)in_sizes; (void)n_in; (void)out_size; (void)ws_size;

  const float* x      = (const float*)d_in[0];  // [1,8,1024,768]
  const float* w_attn = (const float*)d_in[1];  // [768,2304]
  const float* b_attn = (const float*)d_in[2];  // [2304]
  const float* w_proj = (const float*)d_in[3];  // [768,768]
  const float* b_proj = (const float*)d_in[4];  // [768]
  float* out = (float*)d_out;                   // [8192,768] f32

  // Workspace layout (all offsets 256B aligned):
  char* ws = (char*)d_ws;
  _Float16* x_h   = (_Float16*)(ws);             // 8192*768   f16
  _Float16* waT_h = (_Float16*)(ws + 12582912);  // 2304*768   f16 (w_attn^T)
  _Float16* wpT_h = (_Float16*)(ws + 16121856);  //  768*768   f16 (w_proj^T)
  _Float16* qkv_h = (_Float16*)(ws + 17301504);  // 8192*2304  f16
  _Float16* y_h   = (_Float16*)(ws + 55050240);  // 8192*768   f16
  _Float16* vt_h  = (_Float16*)(ws + 67633152);  // [B,H,64,T] f16

  cvt_f32_f16<<<4096, 256, 0, stream>>>(x, x_h, BATCH * SEQ_T * CDIM);
  transpose_cvt_f32_f16<<<2048, 256, 0, stream>>>(w_attn, waT_h, CDIM, QKV_LD);
  transpose_cvt_f32_f16<<<1024, 256, 0, stream>>>(w_proj, wpT_h, CDIM, CDIM);

  // QKV projection: [8192,768] x [768,2304] + b_attn -> f16
  gemm16x64_wmma<false><<<(512 * 36) / 4, 128, 0, stream>>>(
      x_h, waT_h, b_attn, qkv_h, nullptr, QKV_LD, CDIM, 36);

  // V -> [B,H,D,T] for key-contiguous PV fragments
  transpose_v<<<3072, 256, 0, stream>>>(qkv_h, vt_h);

  // Causal flash attention over 96 heads
  flash_attn_wmma<<<(BATCH * H_HEADS * 64) / 4, 128, 0, stream>>>(qkv_h, vt_h,
                                                                  y_h);

  // Output projection: [8192,768] x [768,768] + b_proj -> f32 out
  gemm16x64_wmma<true><<<(512 * 12) / 4, 128, 0, stream>>>(
      y_h, wpT_h, b_proj, nullptr, out, CDIM, CDIM, 12);
}
